// TiSSDLayer_7713761264331
// MI455X (gfx1250) — matrix operand order repaired and checked by reading, not compiled, not run on hardware
//
#include <hip/hip_runtime.h>
#include <cstdint>
#include <cstddef>

// ---------------- problem constants ----------------
#define B_      2
#define SEQ_    2048
#define DMODEL  1024
#define DSTATE  128
#define DCONV   4
#define HEADDIM 64
#define CHUNKS  256
#define DINNER  2048
#define NH_     32
#define CONVDIM 2304     // DINNER + 2*DSTATE
#define DINPROJ 4384     // 2*DINNER + 2*DSTATE + NH
#define DFF     4096
#define NC      8        // SEQ/CHUNK
#define TOK     4096     // B*SEQ

typedef __bf16 bf16;
typedef __attribute__((ext_vector_type(16))) __bf16 bf16x16;
typedef __attribute__((ext_vector_type(8)))  __bf16 bf16x8;
typedef __attribute__((ext_vector_type(8)))  float  f32x8;

#define WMMA_BF16(a,b,c) \
  __builtin_amdgcn_wmma_f32_16x16x32_bf16(false,(a),false,(b),(short)0,(c),false,false)

static __device__ __forceinline__ bf16x16 pack_frag(bf16x8 lo, bf16x8 hi){
  bf16x16 r;
#pragma unroll
  for (int i = 0; i < 8; ++i){ r[i] = lo[i]; r[i+8] = hi[i]; }
  return r;
}
// A fragment (16x32, M=lane16): elems 0..7 -> k0+8*hf+i ; 8..15 -> k0+16+8*hf+i
static __device__ __forceinline__ bf16x16 load_a_frag(const bf16* base, int lda, int m, int k0, int hf){
  const bf16* p0 = base + (size_t)m * lda + k0 + 8*hf;
  return pack_frag(*(const bf16x8*)p0, *(const bf16x8*)(p0 + 16));
}
// B fragment (32x16, N=lane16 of [N,K] buffer): elems 0..7 -> k0+8*hf+i ; 8..15 -> k0+16+8*hf+i
static __device__ __forceinline__ bf16x16 load_b_frag(const bf16* base, int ldb, int n, int k0, int hf){
  const bf16* p0 = base + (size_t)n * ldb + k0 + 8*hf;
  return pack_frag(*(const bf16x8*)p0, *(const bf16x8*)(p0 + 16));
}

static __device__ __forceinline__ float softplusf(float x){
  return (x > 20.f) ? x : log1pf(__expf(x));
}
static __device__ __forceinline__ float siluf(float x){
  return x / (1.f + __expf(-x));
}

// ---------------- generic batched bf16 WMMA GEMM ----------------
// C[M,N] = A[M,K] * BT[N,K]^T (+bias) ; block tile 128x64, 8 waves x (2x2 tiles)
__global__ __launch_bounds__(256) void gemm_bf16_kernel(
    const bf16* __restrict__ A, const bf16* __restrict__ BT,
    float* __restrict__ C, bf16* __restrict__ Cb,
    const float* __restrict__ bias,
    int N, int K, int lda, int ldb, int ldc,
    long long sA, long long sB, long long sC, int act)
{
  int z = blockIdx.z;
  A  += (size_t)z * sA;
  BT += (size_t)z * sB;
  size_t co = (size_t)z * sC;
  int tid = threadIdx.x, lane = tid & 31, wave = tid >> 5;
  int hf = (lane >> 4) & 1, l16 = lane & 15;
  int m0 = blockIdx.x * 128 + (wave & 3) * 32;
  int n0 = blockIdx.y * 64  + (wave >> 2) * 32;

  f32x8 acc[2][2] = {};
  for (int k0 = 0; k0 < K; k0 += 32){
    if (k0 + 128 < K){
      // pull next tiles toward the core 256B ahead (global_prefetch_b8)
      __builtin_prefetch(A  + (size_t)(m0 +      l16) * lda + k0 + 128, 0, 1);
      __builtin_prefetch(A  + (size_t)(m0 + 16 + l16) * lda + k0 + 128, 0, 1);
      __builtin_prefetch(BT + (size_t)(n0 +      l16) * ldb + k0 + 128, 0, 1);
      __builtin_prefetch(BT + (size_t)(n0 + 16 + l16) * ldb + k0 + 128, 0, 1);
    }
    bf16x16 a0 = load_a_frag(A, lda, m0 +      l16, k0, hf);
    bf16x16 a1 = load_a_frag(A, lda, m0 + 16 + l16, k0, hf);
    bf16x16 b0 = load_b_frag(BT, ldb, n0 +      l16, k0, hf);
    bf16x16 b1 = load_b_frag(BT, ldb, n0 + 16 + l16, k0, hf);
    acc[0][0] = WMMA_BF16(a0, b0, acc[0][0]);
    acc[0][1] = WMMA_BF16(a0, b1, acc[0][1]);
    acc[1][0] = WMMA_BF16(a1, b0, acc[1][0]);
    acc[1][1] = WMMA_BF16(a1, b1, acc[1][1]);
  }
#pragma unroll
  for (int tm = 0; tm < 2; ++tm){
#pragma unroll
    for (int tn = 0; tn < 2; ++tn){
      int n = n0 + tn*16 + l16;
      if (n >= N) continue;
      float bv = bias ? bias[n] : 0.f;
#pragma unroll
      for (int r = 0; r < 8; ++r){
        int m = m0 + tm*16 + 8*hf + r;
        float v = acc[tm][tn][r] + bv;
        if (act == 1){ // hardswish
          float u = v + 3.f; u = u < 0.f ? 0.f : (u > 6.f ? 6.f : u);
          v = v * u * (1.f/6.f);
        }
        size_t idx = co + (size_t)m * ldc + n;
        if (C)  C[idx]  = v;
        if (Cb) Cb[idx] = (bf16)v;
      }
    }
  }
}

// ---------------- conversions ----------------
__global__ void f32_to_bf16_kernel(const float* __restrict__ in, bf16* __restrict__ out, size_t n){
  size_t i = (size_t)blockIdx.x * 256 + threadIdx.x;
  if (i < n) out[i] = (bf16)in[i];
}
// W[K,N] f32 -> WT[Npad,K] bf16 (zero padded rows)
__global__ void transpose_w_kernel(const float* __restrict__ W, bf16* __restrict__ WT,
                                   int K, int N, int Npad){
  size_t i = (size_t)blockIdx.x * 256 + threadIdx.x;
  size_t total = (size_t)Npad * K;
  if (i >= total) return;
  int k = (int)(i % K);
  int n = (int)(i / K);
  WT[i] = (n < N) ? (bf16)W[(size_t)k * N + n] : (bf16)0.0f;
}

// ---------------- depthwise conv + silu + dt / dA, operand staging ----------------
__global__ __launch_bounds__(256) void conv_dt_kernel(
    const float* __restrict__ zx, const float* __restrict__ conv_w, const float* __restrict__ conv_b,
    const float* __restrict__ dt_bias, const float* __restrict__ time_decay,
    const float* __restrict__ time_diff, const float* __restrict__ A_log,
    float* __restrict__ xs, bf16* __restrict__ Bmb, bf16* __restrict__ Cmb,
    bf16* __restrict__ BmT, bf16* __restrict__ xdtT, float* __restrict__ dA)
{
  int t = blockIdx.x;            // token 0..TOK-1
  int b = t / SEQ_, l = t % SEQ_;
  int c_chunk = l / CHUNKS, s = l % CHUNKS;
  const float* row = zx + (size_t)t * DINPROJ;
  for (int c = threadIdx.x; c < CONVDIM; c += 256){
    float acc = conv_b[c];
#pragma unroll
    for (int k = 0; k < DCONV; ++k){
      int lk = l - (DCONV - 1) + k;
      if (lk >= 0)
        acc += zx[((size_t)b * SEQ_ + lk) * DINPROJ + DINNER + c] * conv_w[c * DCONV + k];
    }
    float sv = siluf(acc);
    if (c < DINNER){
      xs[(size_t)t * DINNER + c] = sv;
      int h = c >> 6, p = c & 63;
      float dtv = softplusf(row[DINPROJ - NH_ + h] + dt_bias[h]);
      xdtT[ (((size_t)(b * NC + c_chunk) * NH_ + h) * HEADDIM + p) * CHUNKS + s ] = (bf16)(sv * dtv);
    } else if (c < DINNER + DSTATE){
      int n = c - DINNER;
      bf16 v = (bf16)sv;
      Bmb[(size_t)t * DSTATE + n] = v;
      BmT[ ((size_t)(b * NC + c_chunk) * DSTATE + n) * CHUNKS + s ] = v;
    } else {
      int n = c - DINNER - DSTATE;
      Cmb[(size_t)t * DSTATE + n] = (bf16)sv;
    }
  }
  if (threadIdx.x < NH_){
    int h = threadIdx.x;
    float dtv = softplusf(row[DINPROJ - NH_ + h] + dt_bias[h]);
    float Av  = -__expf(A_log[h]);
    float td  = softplusf(time_decay[h]);
    dA[ ((size_t)b * NH_ + h) * SEQ_ + l ] = dtv * Av - td * time_diff[t];
  }
}

// ---------------- per-chunk inclusive cumsum of dA (256 elems per block) ----------------
__global__ __launch_bounds__(256) void cumsum_kernel(const float* __restrict__ dA, float* __restrict__ Acum){
  __shared__ float sh[256];
  int tid = threadIdx.x;
  size_t base = (size_t)blockIdx.x * 256;
  sh[tid] = dA[base + tid];
  __syncthreads();
  for (int off = 1; off < 256; off <<= 1){
    float v = (tid >= off) ? sh[tid - off] : 0.f;
    __syncthreads();
    sh[tid] += v;
    __syncthreads();
  }
  Acum[base + tid] = sh[tid];
}

// ---------------- per-(b,c,h) chunk states: states[p,n] = sum_l xdt[p,l]*decay[l]*B[l,n] ----------------
__global__ __launch_bounds__(256) void states_kernel(
    const bf16* __restrict__ xdtT, const bf16* __restrict__ BmT,
    const float* __restrict__ Acum, float* __restrict__ states)
{
  int bch = blockIdx.x;
  int h = bch & 31, bc = bch >> 5, b = bc >> 3, c = bc & 7;
  __shared__ float acS[256];
  __shared__ float dec[256];
  int tid = threadIdx.x;
  const float* ac = Acum + ((size_t)b * NH_ + h) * SEQ_ + c * CHUNKS;
  acS[tid] = ac[tid];
  __syncthreads();
  dec[tid] = __expf(acS[255] - acS[tid]);
  __syncthreads();

  int lane = tid & 31, wave = tid >> 5;
  int hf = (lane >> 4) & 1, l16 = lane & 15;
  const bf16* Abase = xdtT + (size_t)bch * HEADDIM * CHUNKS;   // [p][l]
  const bf16* Bbase = BmT  + (size_t)bc  * DSTATE  * CHUNKS;   // [n][l]
  int m0 = (wave & 3) * 16;            // p tile
  int n0 = (wave >> 2) * 64;           // 4 n tiles

  f32x8 acc[4] = {};
  for (int k0 = 0; k0 < CHUNKS; k0 += 32){
    // decay-scaled A fragment
    bf16x16 a;
    {
      int m = m0 + l16;
      const bf16* p0 = Abase + (size_t)m * CHUNKS + k0 + 8*hf;
      bf16x8 lo = *(const bf16x8*)p0;
      bf16x8 hi = *(const bf16x8*)(p0 + 16);
#pragma unroll
      for (int i = 0; i < 8; ++i){
        a[i]     = (bf16)((float)lo[i] * dec[k0 +      8*hf + i]);
        a[i + 8] = (bf16)((float)hi[i] * dec[k0 + 16 + 8*hf + i]);
      }
    }
#pragma unroll
    for (int tn = 0; tn < 4; ++tn){
      bf16x16 bb = load_b_frag(Bbase, CHUNKS, n0 + tn*16 + l16, k0, hf);
      acc[tn] = WMMA_BF16(a, bb, acc[tn]);
    }
  }
  float* out = states + (size_t)bch * HEADDIM * DSTATE;
#pragma unroll
  for (int tn = 0; tn < 4; ++tn){
    int n = n0 + tn*16 + l16;
#pragma unroll
    for (int r = 0; r < 8; ++r){
      int p = m0 + 8*hf + r;
      out[(size_t)p * DSTATE + n] = acc[tn][r];
    }
  }
}

// ---------------- inter-chunk scan (sequential over 8 chunks) ----------------
__global__ __launch_bounds__(256) void scan_kernel(
    const float* __restrict__ states, const float* __restrict__ Acum, bf16* __restrict__ prevb)
{
  int bh = blockIdx.x;                 // b*NH + h
  int b = bh >> 5, h = bh & 31;
  int tid = threadIdx.x;
  float S[32];
#pragma unroll
  for (int j = 0; j < 32; ++j) S[j] = 0.f;
  for (int c = 0; c < NC; ++c){
    size_t base = ((size_t)(b * NC + c) * NH_ + h) * (HEADDIM * DSTATE);
    float cd = __expf(Acum[((size_t)b * NH_ + h) * SEQ_ + c * CHUNKS + (CHUNKS - 1)]);
#pragma unroll
    for (int j = 0; j < 32; ++j){
      int e = tid + j * 256;
      prevb[base + e] = (bf16)S[j];
      S[j] = S[j] * cd + states[base + e];
    }
  }
}

// ---------------- Y = Y_diag + Y_off + D*xh for one (b,c,h) ----------------
#define MS_STRIDE 40
#define XB_STRIDE 264   // 256 + 8 bf16 pad: rows land 4 banks apart -> conflict-free b128 reads
__global__ __launch_bounds__(256) void ydiag_kernel(
    const bf16* __restrict__ xdtT, const bf16* __restrict__ Cmb, const bf16* __restrict__ prevb,
    const float* __restrict__ G, const float* __restrict__ Acum,
    const float* __restrict__ Dp, const float* __restrict__ xs, float* __restrict__ Y)
{
  int bch = blockIdx.x;
  int h = bch & 31, bc = bch >> 5, b = bc >> 3, c = bc & 7;
  __shared__ float acS[256];
  __shared__ bf16 Ms[256 * MS_STRIDE];
  __shared__ bf16 xbS[HEADDIM * XB_STRIDE];
  int tid = threadIdx.x;
  const bf16* xb = xdtT + (size_t)bch * HEADDIM * CHUNKS;  // [p][l]

  // --- async-stage the whole 64x256 xdt^T tile into LDS (ASYNCcnt path) ---
  {
    uint32_t ldsbase = (uint32_t)(uintptr_t)(&xbS[0]);
    for (int i = tid; i < HEADDIM * (CHUNKS / 8); i += 256){   // 2048 b128 transfers
      int row = i >> 5, col = i & 31;                           // col: 8-elem groups
      uint32_t laddr = ldsbase + (uint32_t)(row * XB_STRIDE + col * 8) * 2u;
      uint64_t gaddr = (uint64_t)(uintptr_t)(xb + (size_t)row * CHUNKS + col * 8);
      asm volatile("global_load_async_to_lds_b128 %0, %1, off"
                   :: "v"(laddr), "v"(gaddr) : "memory");
    }
  }
  acS[tid] = Acum[((size_t)b * NH_ + h) * SEQ_ + c * CHUNKS + tid];
  asm volatile("s_wait_asynccnt 0x0" ::: "memory");
  __syncthreads();

  int lane = tid & 31, wave = tid >> 5;
  int hf = (lane >> 4) & 1, l16 = lane & 15;
  const float* Grow = G     + (size_t)bc * CHUNKS * CHUNKS;    // [l][s]
  const bf16*  pv   = prevb + (size_t)bch * HEADDIM * DSTATE;  // [p][n]

  f32x8 acc[2][4] = {};
  // phase 1: Y_diag via masked/decayed G slices in LDS
  for (int ks = 0; ks < CHUNKS; ks += 32){
    {
      int l = tid; float al = acS[l];
      const float* g = Grow + (size_t)l * CHUNKS + ks;
#pragma unroll
      for (int sl = 0; sl < 32; ++sl){
        int sidx = ks + sl;
        float v = (sidx <= l) ? __expf(al - acS[sidx]) * g[sl] : 0.f;
        Ms[l * MS_STRIDE + sl] = (bf16)v;
      }
    }
    __syncthreads();
#pragma unroll
    for (int tm = 0; tm < 2; ++tm){
      int m = wave * 32 + tm * 16 + l16;
      const bf16* p0 = &Ms[m * MS_STRIDE + 8*hf];
      bf16x16 a = pack_frag(*(const bf16x8*)p0, *(const bf16x8*)(p0 + 16));
#pragma unroll
      for (int tn = 0; tn < 4; ++tn){
        bf16x16 bb = load_b_frag(&xbS[0], XB_STRIDE, tn*16 + l16, ks, hf);
        acc[tm][tn] = WMMA_BF16(a, bb, acc[tm][tn]);
      }
    }
    __syncthreads();
  }
  // phase 2: Y_off += (state_decay[l]*C[l,:]) @ prev_states[p,:]^T
  for (int k0 = 0; k0 < DSTATE; k0 += 32){
#pragma unroll
    for (int tm = 0; tm < 2; ++tm){
      int m = wave * 32 + tm * 16 + l16;
      float sd = __expf(acS[m]);
      const bf16* p0 = Cmb + ((size_t)(b * SEQ_ + c * CHUNKS + m)) * DSTATE + k0 + 8*hf;
      bf16x8 lo = *(const bf16x8*)p0;
      bf16x8 hi = *(const bf16x8*)(p0 + 16);
      bf16x16 a;
#pragma unroll
      for (int i = 0; i < 8; ++i){
        a[i]     = (bf16)((float)lo[i] * sd);
        a[i + 8] = (bf16)((float)hi[i] * sd);
      }
#pragma unroll
      for (int tn = 0; tn < 4; ++tn){
        bf16x16 bb = load_b_frag(pv, DSTATE, tn*16 + l16, k0, hf);
        acc[tm][tn] = WMMA_BF16(a, bb, acc[tm][tn]);
      }
    }
  }
  float Dh = Dp[h];
#pragma unroll
  for (int tm = 0; tm < 2; ++tm){
#pragma unroll
    for (int tn = 0; tn < 4; ++tn){
#pragma unroll
      for (int r = 0; r < 8; ++r){
        int m = wave * 32 + tm * 16 + 8*hf + r;
        int p = tn * 16 + l16;
        size_t t = (size_t)b * SEQ_ + c * CHUNKS + m;
        float xv = xs[t * DINNER + h * HEADDIM + p];
        Y[t * DINNER + h * HEADDIM + p] = acc[tm][tn][r] + Dh * xv;
      }
    }
  }
}

// ---------------- y = RMS( Y * silu(z) ) * rms_w -> bf16 ----------------
__global__ __launch_bounds__(256) void gate_rms_kernel(
    const float* __restrict__ Y, const float* __restrict__ zx,
    const float* __restrict__ rms_w, bf16* __restrict__ ybf)
{
  __shared__ float red[256];
  int t = blockIdx.x, tid = threadIdx.x;
  const float* y = Y  + (size_t)t * DINNER;
  const float* z = zx + (size_t)t * DINPROJ;   // z = first DINNER cols
  float vals[8]; float ss = 0.f;
#pragma unroll
  for (int j = 0; j < 8; ++j){
    int i = tid + j * 256;
    float v = y[i] * siluf(z[i]);
    vals[j] = v; ss += v * v;
  }
  red[tid] = ss; __syncthreads();
  for (int o = 128; o > 0; o >>= 1){ if (tid < o) red[tid] += red[tid + o]; __syncthreads(); }
  float r = rsqrtf(red[0] * (1.f / DINNER) + 1e-12f);
#pragma unroll
  for (int j = 0; j < 8; ++j){
    int i = tid + j * 256;
    ybf[(size_t)t * DINNER + i] = (bf16)(vals[j] * r * rms_w[i]);
  }
}

// ---------------- LayerNorm(a + rsd) over DMODEL ----------------
__global__ __launch_bounds__(256) void ln_kernel(
    const float* __restrict__ a, const float* __restrict__ rsd,
    const float* __restrict__ g, const float* __restrict__ bt,
    float* __restrict__ of, bf16* __restrict__ ob)
{
  __shared__ float red[256];
  int t = blockIdx.x, tid = threadIdx.x;
  float v[4]; float s = 0.f;
#pragma unroll
  for (int j = 0; j < 4; ++j){
    int i = tid + j * 256;
    float x = a[(size_t)t * DMODEL + i] + (rsd ? rsd[(size_t)t * DMODEL + i] : 0.f);
    v[j] = x; s += x;
  }
  red[tid] = s; __syncthreads();
  for (int o = 128; o > 0; o >>= 1){ if (tid < o) red[tid] += red[tid + o]; __syncthreads(); }
  float mu = red[0] * (1.f / DMODEL);
  __syncthreads();
  s = 0.f;
#pragma unroll
  for (int j = 0; j < 4; ++j){ float d = v[j] - mu; s += d * d; }
  red[tid] = s; __syncthreads();
  for (int o = 128; o > 0; o >>= 1){ if (tid < o) red[tid] += red[tid + o]; __syncthreads(); }
  float inv = rsqrtf(red[0] * (1.f / DMODEL) + 1e-12f);
#pragma unroll
  for (int j = 0; j < 4; ++j){
    int i = tid + j * 256;
    float o2 = (v[j] - mu) * inv * g[i] + bt[i];
    if (of) of[(size_t)t * DMODEL + i] = o2;
    if (ob) ob[(size_t)t * DMODEL + i] = (bf16)o2;
  }
}

__global__ void copy_td_kernel(const float* __restrict__ td, float* __restrict__ out){
  int i = blockIdx.x * 256 + threadIdx.x;
  if (i < TOK) out[i] = td[i];
}

// ---------------- orchestration ----------------
extern "C" void kernel_launch(void* const* d_in, const int* in_sizes, int n_in,
                              void* d_out, int out_size, void* d_ws, size_t ws_size,
                              hipStream_t stream)
{
  (void)in_sizes; (void)n_in; (void)out_size; (void)ws_size;
  const float* x         = (const float*)d_in[0];
  const float* time_diff = (const float*)d_in[1];
  const float* W_in      = (const float*)d_in[2];
  const float* b_in      = (const float*)d_in[3];
  const float* conv_w    = (const float*)d_in[4];
  const float* conv_b    = (const float*)d_in[5];
  const float* A_log     = (const float*)d_in[6];
  const float* dt_bias   = (const float*)d_in[7];
  const float* Dp        = (const float*)d_in[8];
  const float* time_decay= (const float*)d_in[9];
  const float* rms_w     = (const float*)d_in[10];
  const float* W_out     = (const float*)d_in[11];
  const float* b_out     = (const float*)d_in[12];
  const float* ln_g      = (const float*)d_in[13];
  const float* ln_b      = (const float*)d_in[14];
  const float* fc1_w     = (const float*)d_in[15];
  const float* fc1_b     = (const float*)d_in[16];
  const float* fc2_w     = (const float*)d_in[17];
  const float* fc2_b     = (const float*)d_in[18];
  const float* ln2_g     = (const float*)d_in[19];
  const float* ln2_b     = (const float*)d_in[20];

  char* wp = (char*)d_ws;
  auto alloc = [&](size_t bytes)->void*{
    void* p = (void*)wp;
    wp += (bytes + 255) & ~(size_t)255;
    return p;
  };
  const int NPAD_IN = 4416;  // 4384 padded to multiple of 64

  bf16*  xbf    = (bf16*) alloc((size_t)TOK * DMODEL * 2);
  bf16*  WinT   = (bf16*) alloc((size_t)NPAD_IN * DMODEL * 2);
  bf16*  WoutT  = (bf16*) alloc((size_t)DMODEL * DINNER * 2);
  bf16*  fc1T   = (bf16*) alloc((size_t)DFF * DMODEL * 2);
  bf16*  fc2T   = (bf16*) alloc((size_t)DMODEL * DFF * 2);
  float* zx     = (float*)alloc((size_t)TOK * DINPROJ * 4);
  float* xs     = (float*)alloc((size_t)TOK * DINNER * 4);
  bf16*  Bmb    = (bf16*) alloc((size_t)TOK * DSTATE * 2);
  bf16*  Cmb    = (bf16*) alloc((size_t)TOK * DSTATE * 2);
  bf16*  BmT    = (bf16*) alloc((size_t)B_ * NC * DSTATE * CHUNKS * 2);
  bf16*  xdtT   = (bf16*) alloc((size_t)B_ * NC * NH_ * HEADDIM * CHUNKS * 2);
  float* dAb    = (float*)alloc((size_t)B_ * NH_ * SEQ_ * 4);
  float* Acum   = (float*)alloc((size_t)B_ * NH_ * SEQ_ * 4);
  float* G      = (float*)alloc((size_t)B_ * NC * CHUNKS * CHUNKS * 4);
  float* states = (float*)alloc((size_t)B_ * NC * NH_ * HEADDIM * DSTATE * 4);
  bf16*  prevb  = (bf16*) alloc((size_t)B_ * NC * NH_ * HEADDIM * DSTATE * 2);
  float* Ybuf   = (float*)alloc((size_t)TOK * DINNER * 4);
  bf16*  ybf    = (bf16*) alloc((size_t)TOK * DINNER * 2);
  float* hidden = (float*)alloc((size_t)TOK * DMODEL * 4);
  float* hbuf   = (float*)alloc((size_t)TOK * DMODEL * 4);
  bf16*  hbf    = (bf16*) alloc((size_t)TOK * DMODEL * 2);
  bf16*  hidbf  = (bf16*) alloc((size_t)TOK * DFF * 2);
  float* ffn2   = (float*)alloc((size_t)TOK * DMODEL * 4);

  const int thr = 256;
  // stage bf16 operands
  f32_to_bf16_kernel<<<dim3((TOK*DMODEL + 255)/256), thr, 0, stream>>>(x, xbf, (size_t)TOK*DMODEL);
  transpose_w_kernel<<<dim3(((size_t)NPAD_IN*DMODEL + 255)/256), thr, 0, stream>>>(W_in,  WinT,  DMODEL, DINPROJ, NPAD_IN);
  transpose_w_kernel<<<dim3(((size_t)DMODEL*DINNER + 255)/256), thr, 0, stream>>>(W_out, WoutT, DINNER, DMODEL, DMODEL);
  transpose_w_kernel<<<dim3(((size_t)DFF*DMODEL   + 255)/256), thr, 0, stream>>>(fc1_w, fc1T,  DMODEL, DFF,    DFF);
  transpose_w_kernel<<<dim3(((size_t)DMODEL*DFF   + 255)/256), thr, 0, stream>>>(fc2_w, fc2T,  DFF,    DMODEL, DMODEL);

  // zxbcdt = x @ W_in + b_in
  gemm_bf16_kernel<<<dim3(TOK/128, NPAD_IN/64, 1), thr, 0, stream>>>(
      xbf, WinT, zx, nullptr, b_in, DINPROJ, DMODEL, DMODEL, DMODEL, DINPROJ, 0, 0, 0, 0);

  // conv + silu + dt/dA + operand staging
  conv_dt_kernel<<<dim3(TOK), thr, 0, stream>>>(zx, conv_w, conv_b, dt_bias, time_decay,
                                                time_diff, A_log, xs, Bmb, Cmb, BmT, xdtT, dAb);
  // per-chunk cumsum of dA
  cumsum_kernel<<<dim3(B_*NH_*NC), thr, 0, stream>>>(dAb, Acum);

  // G[bc] = Cc @ Bc^T  (batched 256x256x128)
  gemm_bf16_kernel<<<dim3(2, 4, B_*NC), thr, 0, stream>>>(
      Cmb, Bmb, G, nullptr, nullptr, CHUNKS, DSTATE, DSTATE, DSTATE, CHUNKS,
      (long long)CHUNKS*DSTATE, (long long)CHUNKS*DSTATE, (long long)CHUNKS*CHUNKS, 0);

  // chunk states + inter-chunk scan
  states_kernel<<<dim3(B_*NC*NH_), thr, 0, stream>>>(xdtT, BmT, Acum, states);
  scan_kernel<<<dim3(B_*NH_), thr, 0, stream>>>(states, Acum, prevb);

  // Y = Y_diag + Y_off + D*xh
  ydiag_kernel<<<dim3(B_*NC*NH_), thr, 0, stream>>>(xdtT, Cmb, prevb, G, Acum, Dp, xs, Ybuf);

  // gate + RMS norm -> bf16
  gate_rms_kernel<<<dim3(TOK), thr, 0, stream>>>(Ybuf, zx, rms_w, ybf);

  // hidden = y @ W_out + b_out
  gemm_bf16_kernel<<<dim3(TOK/128, DMODEL/64, 1), thr, 0, stream>>>(
      ybf, WoutT, hidden, nullptr, b_out, DMODEL, DINNER, DINNER, DINNER, DMODEL, 0, 0, 0, 0);

  // h = LN(hidden + x)
  ln_kernel<<<dim3(TOK), thr, 0, stream>>>(hidden, x, ln_g, ln_b, hbuf, hbf);

  // hid = hardswish(h @ fc1 + b) -> bf16 only
  gemm_bf16_kernel<<<dim3(TOK/128, DFF/64, 1), thr, 0, stream>>>(
      hbf, fc1T, nullptr, hidbf, fc1_b, DFF, DMODEL, DMODEL, DMODEL, DFF, 0, 0, 0, 1);

  // ffn2 = hid @ fc2 + b
  gemm_bf16_kernel<<<dim3(TOK/128, DMODEL/64, 1), thr, 0, stream>>>(
      hidbf, fc2T, ffn2, nullptr, fc2_b, DMODEL, DFF, DFF, DFF, DMODEL, 0, 0, 0, 0);

  // out = LN(ffn2 + h)  -> d_out[0 : TOK*DMODEL]
  ln_kernel<<<dim3(TOK), thr, 0, stream>>>(ffn2, hbuf, ln2_g, ln2_b, (float*)d_out, nullptr);

  // second tuple element: time_diff passthrough
  copy_td_kernel<<<dim3((TOK + 255)/256), thr, 0, stream>>>(time_diff, (float*)d_out + (size_t)TOK*DMODEL);
}
